// ATTPredictor_52905407152370
// MI455X (gfx1250) — matrix operand order repaired and checked by reading, not compiled, not run on hardware
//
#include <hip/hip_runtime.h>
#include <math.h>

typedef __bf16 bf16;
typedef __attribute__((ext_vector_type(16))) __bf16 v16bf;
typedef __attribute__((ext_vector_type(8)))  __bf16 v8bf;
typedef __attribute__((ext_vector_type(8)))  float  v8f;

#define NROI 512
#define CDIM 256
#define VOC  97
#define TDEC 25
#define TENC 32

__device__ __forceinline__ float sigf(float x) { return 1.0f / (1.0f + expf(-x)); }

// Fragment union: two 16B halves of a 16-element bf16 WMMA operand.
union frag16 {
  v16bf v;
  v8bf  h[2];
  uint4 q[2];
};

// A 16x32 bf16: lane holds K run [half*16, half*16+16)
//  -> two contiguous 16B loads.
__device__ __forceinline__ v16bf load_a_frag(const bf16* __restrict__ row,
                                             int kc, int half) {
  frag16 f;
  f.h[0] = *(const v8bf*)(row + kc + half * 16);
  f.h[1] = *(const v8bf*)(row + kc + half * 16 + 8);
  return f.v;
}

// B 32x16 bf16: lane holds K run [half*16, half*16+16) -> two 16B loads.
__device__ __forceinline__ v16bf load_b_frag(const bf16* __restrict__ row,
                                             int kc, int half) {
  frag16 f;
  f.h[0] = *(const v8bf*)(row + kc + half * 16);
  f.h[1] = *(const v8bf*)(row + kc + half * 16 + 8);
  return f.v;
}

// ---------------------------------------------------------------------------
// Generic WMMA GEMM: C[M,N] = act( A[M,K](bf16) * W[N,K]^T(bf16) + bias )
// One wave per 16x16 tile. K mult of 32, M mult of 16, N ragged (store-guarded;
// loads use a clamped row so dead columns never branch).
// ---------------------------------------------------------------------------
__global__ __launch_bounds__(32) void gemm_bf16_wt(
    const bf16* __restrict__ A, const bf16* __restrict__ W,
    const float* __restrict__ bias, float* __restrict__ Cf,
    bf16* __restrict__ Cb, int M, int N, int K, int act)
{
  const int m0 = blockIdx.x * 16;
  const int n0 = blockIdx.y * 16;
  const int lane = threadIdx.x;
  const int half = lane >> 4;
  const int mrow = lane & 15;
  const int ncol = lane & 15;
  const int n = n0 + ncol;
  const bf16* __restrict__ arow = A + (size_t)(m0 + mrow) * K;
  const bf16* __restrict__ wrow = W + (size_t)(n < N ? n : N - 1) * K;
  v8f acc = {};
#pragma unroll 4
  for (int kc = 0; kc < K; kc += 32) {
    v16bf a = load_a_frag(arow, kc, half);
    v16bf b = load_b_frag(wrow, kc, half);
    acc = __builtin_amdgcn_wmma_f32_16x16x32_bf16(false, a, false, b,
                                                  (short)0, acc, false, false);
  }
#pragma unroll
  for (int r = 0; r < 8; ++r) {
    int m = m0 + r + half * 8;  // C/D: lane -> N, vgpr r -> M (+8 for hi lanes)
    if (n < N) {
      float v = acc[r] + (bias ? bias[n] : 0.0f);
      if (act == 1) v = fmaxf(v, 0.0f);
      size_t idx = (size_t)m * N + n;
      if (Cf) Cf[idx] = v;
      if (Cb) Cb[idx] = (bf16)v;
    }
  }
}

// ---------------------------------------------------------------------------
// Implicit-GEMM 3x3 conv, stride (2,1), pad 1, Cin=Cout=256, W=32.
// Input is channels-last bf16 (n, Hin, 32, 256); weights repacked (co, tap, ci)
// so both fragments are contiguous b128 loads. Output raw f32 NCHW.
// ---------------------------------------------------------------------------
__global__ __launch_bounds__(32) void conv_wmma(
    const bf16* __restrict__ in_cl, const bf16* __restrict__ w2,
    float* __restrict__ out, int Hin, int Hout)
{
  const int n    = blockIdx.x;
  const int co0  = blockIdx.y * 16;
  const int p0   = blockIdx.z * 16;
  const int lane = threadIdx.x;
  const int half = lane >> 4;
  const int mrow = lane & 15;
  const int ncol = lane & 15;
  const int p  = p0 + ncol;
  const int ho = p >> 5, wo = p & 31;
  const bf16* __restrict__ arow0 = w2 + (size_t)(co0 + mrow) * (9 * CDIM);
  v8f acc = {};
  for (int tap = 0; tap < 9; ++tap) {
    const int kh = tap / 3, kw = tap % 3;
    const int hi = ho * 2 + kh - 1;
    const int wi = wo + kw - 1;
    const bool inb = (hi >= 0) && (hi < Hin) && (wi >= 0) && (wi < 32);
    const bf16* __restrict__ arow = arow0 + tap * CDIM;
    const bf16* __restrict__ brow =
        in_cl + (((size_t)n * Hin + (inb ? hi : 0)) * 32 + (inb ? wi : 0)) * CDIM;
#pragma unroll
    for (int c0 = 0; c0 < CDIM; c0 += 32) {
      v16bf a = load_a_frag(arow, c0, half);
      frag16 b;
      if (inb) {
        b.h[0] = *(const v8bf*)(brow + c0 + half * 16);
        b.h[1] = *(const v8bf*)(brow + c0 + half * 16 + 8);
      } else {
        b.q[0] = uint4{0, 0, 0, 0};
        b.q[1] = uint4{0, 0, 0, 0};
      }
      acc = __builtin_amdgcn_wmma_f32_16x16x32_bf16(false, a, false, b.v,
                                                    (short)0, acc, false, false);
    }
  }
  const int HW = Hout * 32;
#pragma unroll
  for (int r = 0; r < 8; ++r) {
    int co = co0 + r + half * 8;
    out[((size_t)n * CDIM + co) * HW + p] = acc[r];
  }
}

// ---------------------------------------------------------------------------
// Repack conv weights (co, ci, kh, kw) f32 -> (co, tap, ci) bf16.
// ---------------------------------------------------------------------------
__global__ void repack_conv_w(const float* __restrict__ w, bf16* __restrict__ o)
{
  int idx = blockIdx.x * 256 + threadIdx.x;      // (co*9 + tap)*256 + ci
  if (idx >= CDIM * 9 * CDIM) return;
  int ci = idx & 255;
  int tap = (idx >> 8) % 9;
  int co = idx / (9 * CDIM);
  o[idx] = (bf16)w[(size_t)co * (CDIM * 9) + (size_t)ci * 9 + tap];
}

// rois NCHW f32 -> channels-last (n, 8, 32, 256) bf16
__global__ void repack_in_cl(const float* __restrict__ x, bf16* __restrict__ o)
{
  int idx = blockIdx.x * 256 + threadIdx.x;      // ((n*8+h)*32+w)*256+c
  if (idx >= NROI * 8 * 32 * CDIM) return;
  int c = idx & 255;
  int w = (idx >> 8) & 31;
  int h = (idx >> 13) & 7;
  int n = idx >> 16;
  o[idx] = (bf16)x[(((size_t)n * CDIM + c) * 8 + h) * 32 + w];
}

// ---------------------------------------------------------------------------
// GroupNorm stats: one block per (n, group of 8 channels).
// ---------------------------------------------------------------------------
__global__ void gn_stats(const float* __restrict__ x, float* __restrict__ mu,
                         float* __restrict__ rstd, int HW)
{
  const int n = blockIdx.x >> 5, g = blockIdx.x & 31;
  const int cnt = 8 * HW;
  const float* base = x + ((size_t)n * CDIM + g * 8) * HW;
  float s = 0.f, s2 = 0.f;
  for (int i = threadIdx.x; i < cnt; i += 256) {
    float v = base[i]; s += v; s2 += v * v;
  }
  __shared__ float ss[256], qq[256];
  ss[threadIdx.x] = s; qq[threadIdx.x] = s2; __syncthreads();
  for (int st = 128; st; st >>= 1) {
    if (threadIdx.x < st) { ss[threadIdx.x] += ss[threadIdx.x + st];
                            qq[threadIdx.x] += qq[threadIdx.x + st]; }
    __syncthreads();
  }
  if (!threadIdx.x) {
    float m = ss[0] / cnt;
    float var = qq[0] / cnt - m * m;
    mu[blockIdx.x] = m;
    rstd[blockIdx.x] = rsqrtf(var + 1e-5f);
  }
}

// GN + ReLU, NCHW f32 in -> channels-last bf16 out (conv2's WMMA input).
__global__ void gn_apply_cl(const float* __restrict__ x, const float* __restrict__ mu,
                            const float* __restrict__ rstd, const float* __restrict__ sc,
                            const float* __restrict__ bi, bf16* __restrict__ y,
                            int HW, int total)
{
  int idx = blockIdx.x * 256 + threadIdx.x;
  if (idx >= total) return;
  int hw = idx % HW;
  int c = (idx / HW) & (CDIM - 1);
  int n = idx / (HW * CDIM);
  int g = c >> 3;
  float v = (x[idx] - mu[n * 32 + g]) * rstd[n * 32 + g] * sc[c] + bi[c];
  y[((size_t)n * HW + hw) * CDIM + c] = (bf16)fmaxf(v, 0.0f);
}

// GN2 + ReLU + mean over H(=2) + transpose -> x_bf[t][n][c]
__global__ void x_build(const float* __restrict__ raw, const float* __restrict__ mu,
                        const float* __restrict__ rstd, const float* __restrict__ sc,
                        const float* __restrict__ bi, bf16* __restrict__ xbf)
{
  int idx = blockIdx.x * 256 + threadIdx.x;
  if (idx >= NROI * CDIM * TENC) return;
  int t = idx & 31;
  int c = (idx >> 5) & (CDIM - 1);
  int n = idx >> 13;
  int g = c >> 3;
  float m = mu[n * 32 + g], rs = rstd[n * 32 + g];
  size_t b0 = (((size_t)n * CDIM + c) * 2 + 0) * 32 + t;
  float v0 = fmaxf((raw[b0]      - m) * rs * sc[c] + bi[c], 0.0f);
  float v1 = fmaxf((raw[b0 + 32] - m) * rs * sc[c] + bi[c], 0.0f);
  xbf[((size_t)t * NROI + n) * CDIM + c] = (bf16)(0.5f * (v0 + v1));
}

// ---------------------------------------------------------------------------
// LSTM pointwise (gate order i,f,g,o). Gin holds x@Wih^T + bih + bhh.
// ---------------------------------------------------------------------------
__global__ void lstm_point(const float* __restrict__ Gin, const float* __restrict__ Ghh,
                           float* __restrict__ cst, bf16* __restrict__ h_bf,
                           bf16* __restrict__ hcat, int t, int dirOff)
{
  int idx = blockIdx.x * 256 + threadIdx.x;   // < 512*256
  int n = idx >> 8, c = idx & 255;
  size_t g = (size_t)n * 1024;
  float gi = Gin[g + c]       + Ghh[g + c];
  float gf = Gin[g + 256 + c] + Ghh[g + 256 + c];
  float gg = Gin[g + 512 + c] + Ghh[g + 512 + c];
  float go = Gin[g + 768 + c] + Ghh[g + 768 + c];
  float cn = sigf(gf) * cst[idx] + sigf(gi) * tanhf(gg);
  float h  = sigf(go) * tanhf(cn);
  cst[idx] = cn;
  h_bf[idx] = (bf16)h;
  hcat[((size_t)t * NROI + n) * (2 * CDIM) + dirOff + c] = (bf16)h;
}

// ---------------------------------------------------------------------------
// Attention: score[te,n] = vat_b + sum_c vat_w[c]*tanh(h[n,c]+enc[te,n,c])
// ---------------------------------------------------------------------------
__global__ void att_score(const float* __restrict__ h, const float* __restrict__ enc,
                          const float* __restrict__ vw, const float* __restrict__ vb,
                          float* __restrict__ scores)
{
  int row = blockIdx.x * 8 + (threadIdx.x >> 5);   // row = te*512 + n
  int lane = threadIdx.x & 31;
  int n = row & (NROI - 1);
  const float* e  = enc + (size_t)row * CDIM;
  const float* hh = h + (size_t)n * CDIM;
  float s = 0.f;
#pragma unroll
  for (int j = 0; j < 8; ++j) {
    int c = lane + j * 32;
    s += vw[c] * tanhf(hh[c] + e[c]);
  }
  for (int o = 16; o; o >>= 1) s += __shfl_xor(s, o, 32);
  if (!lane) scores[row] = s + vb[0];
}

// softmax over T_enc, context vector, and decoder-input embedding concat.
__global__ void att_ctx(const float* __restrict__ scores, const float* __restrict__ enc,
                        const float* __restrict__ aemb, const int* __restrict__ targets,
                        bf16* __restrict__ cat, int t)
{
  __shared__ float aw[TENC];
  int n = blockIdx.x;
  if (threadIdx.x < 32) {
    int te = threadIdx.x;
    float s = scores[te * NROI + n];
    float m = s;
    for (int o = 16; o; o >>= 1) m = fmaxf(m, __shfl_xor(m, o, 32));
    float e = expf(s - m);
    float sum = e;
    for (int o = 16; o; o >>= 1) sum += __shfl_xor(sum, o, 32);
    aw[te] = e / sum;
  }
  __syncthreads();
  int c = threadIdx.x;
  float ctx = 0.f;
  for (int te = 0; te < TENC; ++te)
    ctx += aw[te] * enc[((size_t)te * NROI + n) * CDIM + c];
  cat[(size_t)n * (2 * CDIM) + CDIM + c] = (bf16)ctx;
  int tok = (t == 0) ? 0 : targets[n * TDEC + t - 1];
  cat[(size_t)n * (2 * CDIM) + c] = (bf16)aemb[(size_t)tok * CDIM + c];
}

__global__ void gru_point(const float* __restrict__ gi, const float* __restrict__ gh,
                          float* __restrict__ h, bf16* __restrict__ h_bf)
{
  int idx = blockIdx.x * 256 + threadIdx.x;   // < 512*256
  int n = idx >> 8, c = idx & 255;
  size_t g = (size_t)n * 768;
  float r  = sigf(gi[g + c]       + gh[g + c]);
  float z  = sigf(gi[g + 256 + c] + gh[g + 256 + c]);
  float ng = tanhf(gi[g + 512 + c] + r * gh[g + 512 + c]);
  float hn = (1.0f - z) * ng + z * h[idx];
  h[idx] = hn;
  h_bf[idx] = (bf16)hn;
}

// per-row NLL: losses[n] = logsumexp(logits[n,:]) - logits[n, tgt]
__global__ void loss_k(const float* __restrict__ logits, const int* __restrict__ targets,
                       float* __restrict__ losses, int t)
{
  int wid = (blockIdx.x * 256 + threadIdx.x) >> 5;   // exactly 512 waves
  int lane = threadIdx.x & 31;
  const float* row = logits + (size_t)wid * VOC;
  float m = -1e30f;
  for (int j = lane; j < VOC; j += 32) m = fmaxf(m, row[j]);
  for (int o = 16; o; o >>= 1) m = fmaxf(m, __shfl_xor(m, o, 32));
  float s = 0.f;
  for (int j = lane; j < VOC; j += 32) s += expf(row[j] - m);
  for (int o = 16; o; o >>= 1) s += __shfl_xor(s, o, 32);
  if (!lane) {
    int tgt = targets[wid * TDEC + t];
    losses[wid] = (m + logf(s)) - row[tgt];
  }
}

__global__ void final_sum(const float* __restrict__ losses, float* __restrict__ out)
{
  __shared__ float sm[256];
  float s = 0.f;
  for (int i = threadIdx.x; i < TDEC * NROI; i += 256) s += losses[i];
  sm[threadIdx.x] = s; __syncthreads();
  for (int st = 128; st; st >>= 1) {
    if (threadIdx.x < st) sm[threadIdx.x] += sm[threadIdx.x + st];
    __syncthreads();
  }
  if (!threadIdx.x) out[0] = sm[0] * (1.0f / (float)NROI);
}

__global__ void vec_add(const float* __restrict__ a, const float* __restrict__ b,
                        float* __restrict__ o, int nElem)
{
  int i = blockIdx.x * 256 + threadIdx.x;
  if (i < nElem) o[i] = a[i] + b[i];
}

__global__ void f2bf(const float* __restrict__ a, bf16* __restrict__ o, int nElem)
{
  int i = blockIdx.x * 256 + threadIdx.x;
  if (i < nElem) o[i] = (bf16)a[i];
}

__global__ void zero_f32(float* __restrict__ p, int nElem)
{
  int i = blockIdx.x * 256 + threadIdx.x;
  if (i < nElem) p[i] = 0.0f;
}

// ---------------------------------------------------------------------------
extern "C" void kernel_launch(void* const* d_in, const int* in_sizes, int n_in,
                              void* d_out, int out_size, void* d_ws, size_t ws_size,
                              hipStream_t stream)
{
  const float* rois    = (const float*)d_in[0];
  const int*   targets = (const int*)d_in[1];
  const float* conv1_w = (const float*)d_in[2];
  const float* gn1_s   = (const float*)d_in[3];
  const float* gn1_b   = (const float*)d_in[4];
  const float* conv2_w = (const float*)d_in[5];
  const float* gn2_s   = (const float*)d_in[6];
  const float* gn2_b   = (const float*)d_in[7];
  const float* wih_f   = (const float*)d_in[8];
  const float* whh_f   = (const float*)d_in[9];
  const float* bih_f   = (const float*)d_in[10];
  const float* bhh_f   = (const float*)d_in[11];
  const float* wih_b   = (const float*)d_in[12];
  const float* whh_b   = (const float*)d_in[13];
  const float* bih_b   = (const float*)d_in[14];
  const float* bhh_b   = (const float*)d_in[15];
  const float* emb_w   = (const float*)d_in[16];
  const float* emb_b   = (const float*)d_in[17];
  const float* att_e   = (const float*)d_in[18];
  const float* comb_w  = (const float*)d_in[19];
  const float* comb_b  = (const float*)d_in[20];
  const float* gru_wih = (const float*)d_in[21];
  const float* gru_whh = (const float*)d_in[22];
  const float* gru_bih = (const float*)d_in[23];
  const float* gru_bhh = (const float*)d_in[24];
  const float* out_w   = (const float*)d_in[25];
  const float* out_b   = (const float*)d_in[26];
  const float* vat_w   = (const float*)d_in[27];
  const float* vat_b   = (const float*)d_in[28];

  // ---- workspace layout ----
  char* ws = (char*)d_ws;
  size_t off = 0;
  auto alloc = [&](size_t bytes) -> char* {
    off = (off + 255) & ~(size_t)255;
    char* p = ws + off;
    off += bytes;
    return p;
  };
  // region A: conv1 raw f32 (67MB), reused as LSTM input-gates Gin
  float* bigA = (float*)alloc((size_t)NROI * CDIM * 4 * 32 * 4);
  // region B: conv2 raw f32 (33.5MB), reused as hcat (bf16 32*512*512)
  float* bigB = (float*)alloc((size_t)NROI * CDIM * 2 * 32 * 4);
  // region C: rois_cl bf16 (67MB), reused as y1_cl bf16 (33.5MB)
  bf16* bigC  = (bf16*)alloc((size_t)NROI * CDIM * 8 * 32 * 2);
  bf16* x_bf  = (bf16*)alloc((size_t)TENC * NROI * CDIM * 2);
  float* enc  = (float*)alloc((size_t)TENC * NROI * CDIM * 4);
  float* mu   = (float*)alloc(NROI * 32 * 4);
  float* rstd = (float*)alloc(NROI * 32 * 4);
  float* bsum_f = (float*)alloc(4 * CDIM * 4);
  float* bsum_b = (float*)alloc(4 * CDIM * 4);
  float* Ghh    = (float*)alloc((size_t)NROI * 4 * CDIM * 4);
  float* cst    = (float*)alloc((size_t)NROI * CDIM * 4);
  bf16*  h_l_bf = (bf16*)alloc((size_t)NROI * CDIM * 2);
  float* h_dec  = (float*)alloc((size_t)NROI * CDIM * 4);
  bf16*  h_d_bf = (bf16*)alloc((size_t)NROI * CDIM * 2);
  float* scores = (float*)alloc((size_t)TENC * NROI * 4);
  bf16*  cat_bf = (bf16*)alloc((size_t)NROI * 2 * CDIM * 2);
  bf16*  xin_bf = (bf16*)alloc((size_t)NROI * CDIM * 2);
  float* gi     = (float*)alloc((size_t)NROI * 3 * CDIM * 4);
  float* gh     = (float*)alloc((size_t)NROI * 3 * CDIM * 4);
  float* logits = (float*)alloc((size_t)NROI * VOC * 4);
  float* losses = (float*)alloc((size_t)TDEC * NROI * 4);
  // bf16 weight copies
  bf16* c1w_bf  = (bf16*)alloc((size_t)CDIM * 9 * CDIM * 2);
  bf16* c2w_bf  = (bf16*)alloc((size_t)CDIM * 9 * CDIM * 2);
  bf16* wihf_bf = (bf16*)alloc((size_t)4 * CDIM * CDIM * 2);
  bf16* whhf_bf = (bf16*)alloc((size_t)4 * CDIM * CDIM * 2);
  bf16* wihb_bf = (bf16*)alloc((size_t)4 * CDIM * CDIM * 2);
  bf16* whhb_bf = (bf16*)alloc((size_t)4 * CDIM * CDIM * 2);
  bf16* embw_bf = (bf16*)alloc((size_t)CDIM * 2 * CDIM * 2);
  bf16* combw_bf= (bf16*)alloc((size_t)CDIM * 2 * CDIM * 2);
  bf16* gwih_bf = (bf16*)alloc((size_t)3 * CDIM * CDIM * 2);
  bf16* gwhh_bf = (bf16*)alloc((size_t)3 * CDIM * CDIM * 2);
  bf16* outw_bf = (bf16*)alloc((size_t)VOC * CDIM * 2);
  (void)ws_size; (void)in_sizes; (void)n_in; (void)out_size;

  float* conv1_raw = bigA;
  float* Gin       = bigA;          // reuse after conv1 consumed
  float* conv2_raw = bigB;
  bf16*  hcat      = (bf16*)bigB;   // reuse after conv2 consumed
  bf16*  rois_cl   = bigC;
  bf16*  y1_cl     = bigC;          // reuse after conv1 consumed

  #define CDIV(a, b) (((a) + (b) - 1) / (b))

  // ---- weight repack/convert ----
  repack_conv_w<<<CDIV(CDIM * 9 * CDIM, 256), 256, 0, stream>>>(conv1_w, c1w_bf);
  repack_conv_w<<<CDIV(CDIM * 9 * CDIM, 256), 256, 0, stream>>>(conv2_w, c2w_bf);
  f2bf<<<CDIV(4 * CDIM * CDIM, 256), 256, 0, stream>>>(wih_f, wihf_bf, 4 * CDIM * CDIM);
  f2bf<<<CDIV(4 * CDIM * CDIM, 256), 256, 0, stream>>>(whh_f, whhf_bf, 4 * CDIM * CDIM);
  f2bf<<<CDIV(4 * CDIM * CDIM, 256), 256, 0, stream>>>(wih_b, wihb_bf, 4 * CDIM * CDIM);
  f2bf<<<CDIV(4 * CDIM * CDIM, 256), 256, 0, stream>>>(whh_b, whhb_bf, 4 * CDIM * CDIM);
  f2bf<<<CDIV(2 * CDIM * CDIM, 256), 256, 0, stream>>>(emb_w, embw_bf, 2 * CDIM * CDIM);
  f2bf<<<CDIV(2 * CDIM * CDIM, 256), 256, 0, stream>>>(comb_w, combw_bf, 2 * CDIM * CDIM);
  f2bf<<<CDIV(3 * CDIM * CDIM, 256), 256, 0, stream>>>(gru_wih, gwih_bf, 3 * CDIM * CDIM);
  f2bf<<<CDIV(3 * CDIM * CDIM, 256), 256, 0, stream>>>(gru_whh, gwhh_bf, 3 * CDIM * CDIM);
  f2bf<<<CDIV(VOC * CDIM, 256), 256, 0, stream>>>(out_w, outw_bf, VOC * CDIM);
  vec_add<<<CDIV(4 * CDIM, 256), 256, 0, stream>>>(bih_f, bhh_f, bsum_f, 4 * CDIM);
  vec_add<<<CDIV(4 * CDIM, 256), 256, 0, stream>>>(bih_b, bhh_b, bsum_b, 4 * CDIM);

  // ---- conv1 + GN + ReLU (channels-last bf16 out) ----
  repack_in_cl<<<CDIV(NROI * CDIM * 8 * 32, 256), 256, 0, stream>>>(rois, rois_cl);
  conv_wmma<<<dim3(NROI, 16, 8), 32, 0, stream>>>(rois_cl, c1w_bf, conv1_raw, 8, 4);
  gn_stats<<<NROI * 32, 256, 0, stream>>>(conv1_raw, mu, rstd, 128);
  gn_apply_cl<<<CDIV(NROI * CDIM * 128, 256), 256, 0, stream>>>(
      conv1_raw, mu, rstd, gn1_s, gn1_b, y1_cl, 128, NROI * CDIM * 128);

  // ---- conv2 + GN + ReLU + meanH + transpose -> x_bf (t,n,c) ----
  conv_wmma<<<dim3(NROI, 16, 4), 32, 0, stream>>>(y1_cl, c2w_bf, conv2_raw, 4, 2);
  gn_stats<<<NROI * 32, 256, 0, stream>>>(conv2_raw, mu, rstd, 64);
  x_build<<<CDIV(NROI * CDIM * TENC, 256), 256, 0, stream>>>(
      conv2_raw, mu, rstd, gn2_s, gn2_b, x_bf);

  // ---- BiLSTM ----
  for (int dir = 0; dir < 2; ++dir) {
    const bf16* Wih = dir ? wihb_bf : wihf_bf;
    const bf16* Whh = dir ? whhb_bf : whhf_bf;
    const float* bs = dir ? bsum_b : bsum_f;
    gemm_bf16_wt<<<dim3(TENC * NROI / 16, 4 * CDIM / 16), 32, 0, stream>>>(
        x_bf, Wih, bs, Gin, nullptr, TENC * NROI, 4 * CDIM, CDIM, 0);
    zero_f32<<<CDIV(NROI * CDIM, 256), 256, 0, stream>>>(cst, NROI * CDIM);
    zero_f32<<<CDIV(NROI * CDIM / 2, 256), 256, 0, stream>>>((float*)h_l_bf,
                                                             NROI * CDIM / 2);
    for (int s = 0; s < TENC; ++s) {
      int t = dir ? (TENC - 1 - s) : s;
      gemm_bf16_wt<<<dim3(NROI / 16, 4 * CDIM / 16), 32, 0, stream>>>(
          h_l_bf, Whh, nullptr, Ghh, nullptr, NROI, 4 * CDIM, CDIM, 0);
      lstm_point<<<CDIV(NROI * CDIM, 256), 256, 0, stream>>>(
          Gin + (size_t)t * NROI * 4 * CDIM, Ghh, cst, h_l_bf, hcat, t, dir * CDIM);
    }
  }

  // ---- enc = hcat @ emb_w^T + emb_b ----
  gemm_bf16_wt<<<dim3(TENC * NROI / 16, CDIM / 16), 32, 0, stream>>>(
      hcat, embw_bf, emb_b, enc, nullptr, TENC * NROI, CDIM, 2 * CDIM, 0);

  // ---- attention GRU decoder ----
  zero_f32<<<CDIV(NROI * CDIM, 256), 256, 0, stream>>>(h_dec, NROI * CDIM);
  zero_f32<<<CDIV(NROI * CDIM / 2, 256), 256, 0, stream>>>((float*)h_d_bf,
                                                           NROI * CDIM / 2);
  for (int t = 0; t < TDEC; ++t) {
    att_score<<<TENC * NROI / 8, 256, 0, stream>>>(h_dec, enc, vat_w, vat_b, scores);
    att_ctx<<<NROI, 256, 0, stream>>>(scores, enc, att_e, targets, cat_bf, t);
    gemm_bf16_wt<<<dim3(NROI / 16, CDIM / 16), 32, 0, stream>>>(
        cat_bf, combw_bf, comb_b, nullptr, xin_bf, NROI, CDIM, 2 * CDIM, 1);
    gemm_bf16_wt<<<dim3(NROI / 16, 3 * CDIM / 16), 32, 0, stream>>>(
        xin_bf, gwih_bf, gru_bih, gi, nullptr, NROI, 3 * CDIM, CDIM, 0);
    gemm_bf16_wt<<<dim3(NROI / 16, 3 * CDIM / 16), 32, 0, stream>>>(
        h_d_bf, gwhh_bf, gru_bhh, gh, nullptr, NROI, 3 * CDIM, CDIM, 0);
    gru_point<<<CDIV(NROI * CDIM, 256), 256, 0, stream>>>(gi, gh, h_dec, h_d_bf);
    gemm_bf16_wt<<<dim3(NROI / 16, CDIV(VOC, 16)), 32, 0, stream>>>(
        h_d_bf, outw_bf, out_b, logits, nullptr, NROI, VOC, CDIM, 0);
    loss_k<<<NROI / 8, 256, 0, stream>>>(logits, targets, losses + (size_t)t * NROI, t);
  }
  final_sum<<<1, 256, 0, stream>>>(losses, (float*)d_out);
}
